// GCGRUCell_63015760167428
// MI455X (gfx1250) — matrix operand order repaired and checked by reading, not compiled, not run on hardware
//
#include <hip/hip_runtime.h>
#include <hip/hip_bf16.h>
#include <math.h>

// ---------------------------------------------------------------------------
// MRGCN (2 layers) + GRU cell for MI455X (gfx1250), wave32 + WMMA.
//   - activations/weights pre-converted once to f16 (weights transposed) so
//     every WMMA fragment loads as two global_load_b128
//   - GEMM K-loop fully unrolled (K=128 template const) for load pipelining
//   - bias+ReLU+f16-convert fused: h16 produced directly, no f32 h buffer
//   - edge scatter via global_atomic_add_f32 (support/agg live in 192MB L2)
//   - GRU fully fused: 6 WMMA accumulators + gate chain in registers
// ---------------------------------------------------------------------------

typedef __attribute__((ext_vector_type(16))) _Float16 v16h;
typedef __attribute__((ext_vector_type(8)))  _Float16 v8h;
typedef __attribute__((ext_vector_type(4)))  _Float16 v4h;
typedef __attribute__((ext_vector_type(8)))  float    v8f;

__device__ __forceinline__ v8f wmma_f16(v16h a, v16h b, v8f c) {
    // (neg_a, A, neg_b, B, c_mod, C, reuse_a, reuse_b)
    return __builtin_amdgcn_wmma_f32_16x16x32_f16(false, a, false, b, (short)0, c,
                                                  false, false);
}

// A fragment (16x32 f16, row-major A16[lda]): lane(half, m) holds
//   K = half*16 + e (e=0..15) -> contiguous 16 halfs at k0 + half*16
__device__ __forceinline__ v16h load_a16(const _Float16* __restrict__ A, int lda,
                                         int r0, int k0, int m, int half) {
    const _Float16* p = A + (size_t)(r0 + m) * lda + k0 + half * 16;
    v8h lo = *reinterpret_cast<const v8h*>(p);        // 16B load
    v8h hi = *reinterpret_cast<const v8h*>(p + 8);    // 16B load
    return __builtin_shufflevector(lo, hi, 0, 1, 2, 3, 4, 5, 6, 7,
                                   8, 9, 10, 11, 12, 13, 14, 15);
}

// B fragment (32x16 f16) from TRANSPOSED weights Bt[C x K] row-major:
// lane(half, n) holds K = half*16 + e (e=0..15) of original column c0+n,
// i.e. 16 contiguous halfs of row (c0+n) of Bt.
__device__ __forceinline__ v16h load_bt16(const _Float16* __restrict__ Bt, int ldk,
                                          int k0, int c0, int n, int half) {
    const _Float16* p = Bt + (size_t)(c0 + n) * ldk + k0 + half * 16;
    v8h lo = *reinterpret_cast<const v8h*>(p);
    v8h hi = *reinterpret_cast<const v8h*>(p + 8);
    return __builtin_shufflevector(lo, hi, 0, 1, 2, 3, 4, 5, 6, 7,
                                   8, 9, 10, 11, 12, 13, 14, 15);
}

// ---------------------------------------------------------------------------
// f32 -> f16 conversion (vector of 4 per thread)
// ---------------------------------------------------------------------------
__global__ void cvt_f16_kernel(const float* __restrict__ in,
                               _Float16* __restrict__ out, int n) {
    int i = (blockIdx.x * blockDim.x + threadIdx.x) * 4;
    if (i < n) {
        float4 v = *reinterpret_cast<const float4*>(in + i);
        v4h o;
        o[0] = (_Float16)v.x; o[1] = (_Float16)v.y;
        o[2] = (_Float16)v.z; o[3] = (_Float16)v.w;
        *reinterpret_cast<v4h*>(out + i) = o;
    }
}

// W [K x C] row-major  ->  Wt [C x K] f16 row-major
__global__ void transpose_cvt_kernel(const float* __restrict__ W,
                                     _Float16* __restrict__ Wt, int K, int C) {
    int i = blockIdx.x * blockDim.x + threadIdx.x;
    if (i < K * C) {
        int k = i / C, c = i % C;
        Wt[(size_t)c * K + k] = (_Float16)W[i];
    }
}

// ---------------------------------------------------------------------------
// GEMM: C[M x ldc] = A16[M x K] * B (given as Bt16[ldc x K]).
// grid = (M/16, ldc/64), block = 32. Each wave: 16 rows x 64 cols (4 tiles),
// A fragment loaded once per K-step; K-loop fully unrolled for pipelining.
// ---------------------------------------------------------------------------
template <int K>
__global__ void gemm_wmma_kernel(const _Float16* __restrict__ A16,
                                 const _Float16* __restrict__ Bt16,
                                 float* __restrict__ C,
                                 int ldc) {
    const int r0    = blockIdx.x * 16;
    const int cbase = blockIdx.y * 64;
    const int lane  = threadIdx.x & 31;
    const int half  = lane >> 4;
    const int m     = lane & 15;
    const int n     = m;

    v8f acc[4] = {{}, {}, {}, {}};
#pragma unroll
    for (int k0 = 0; k0 < K; k0 += 32) {
        v16h a = load_a16(A16, K, r0, k0, m, half);
#pragma unroll
        for (int j = 0; j < 4; ++j) {
            v16h b = load_bt16(Bt16, K, k0, cbase + j * 16, n, half);
            acc[j] = wmma_f16(a, b, acc[j]);
        }
    }
#pragma unroll
    for (int j = 0; j < 4; ++j) {
#pragma unroll
        for (int v = 0; v < 8; ++v) {
            int row = r0 + v + 8 * half;  // D-matrix: M = v + 8*half, N = lane&15
            C[(size_t)row * ldc + cbase + j * 16 + n] = acc[j][v];
        }
    }
}

// ---------------------------------------------------------------------------
// agg = 0
// ---------------------------------------------------------------------------
__global__ void zero_kernel(float* __restrict__ p, int n) {
    int i = blockIdx.x * blockDim.x + threadIdx.x;
    if (i < n) p[i] = 0.0f;
}

// ---------------------------------------------------------------------------
// Edge scatter: agg[dst] += support[src] * w.  32 threads/edge, float4 each.
// support/agg are 25.6MB each -> atomics resolve in the 192MB L2.
// ---------------------------------------------------------------------------
__global__ void scatter_kernel(float* __restrict__ agg,
                               const float* __restrict__ support,
                               const int* __restrict__ src,
                               const int* __restrict__ dst,
                               const float* __restrict__ w,
                               int E) {
    const int D = 128;
    long idx = (long)blockIdx.x * blockDim.x + threadIdx.x;
    long total = (long)E * 32;  // D/4 chunks per edge
    if (idx >= total) return;
    int e = (int)(idx >> 5);
    int c = (int)(idx & 31) * 4;
    int s = src[e];
    int d = dst[e];
    float wt = w[e];
    const float4 v = *reinterpret_cast<const float4*>(support + (size_t)s * D + c);
    float* base = agg + (size_t)d * D + c;
    atomicAdd(base + 0, v.x * wt);
    atomicAdd(base + 1, v.y * wt);
    atomicAdd(base + 2, v.z * wt);
    atomicAdd(base + 3, v.w * wt);
}

// ---------------------------------------------------------------------------
// h16 = f16(relu(agg + b))  -- fused bias+ReLU+convert, float4/v4h per thread.
// Eliminates the f32 h buffer and an extra full-array conversion pass.
// ---------------------------------------------------------------------------
__global__ void biasrelu_cvt_kernel(_Float16* __restrict__ h16,
                                    const float* __restrict__ agg,
                                    const float* __restrict__ b,
                                    int nd) {
    int i = (blockIdx.x * blockDim.x + threadIdx.x) * 4;
    if (i < nd) {
        float4 v = *reinterpret_cast<const float4*>(agg + i);
        int c = i & 127;  // D = 128, rows are 128-aligned so c..c+3 same row
        float r0 = v.x + b[c + 0];
        float r1 = v.y + b[c + 1];
        float r2 = v.z + b[c + 2];
        float r3 = v.w + b[c + 3];
        v4h o;
        o[0] = (_Float16)(r0 > 0.0f ? r0 : 0.0f);
        o[1] = (_Float16)(r1 > 0.0f ? r1 : 0.0f);
        o[2] = (_Float16)(r2 > 0.0f ? r2 : 0.0f);
        o[3] = (_Float16)(r3 > 0.0f ? r3 : 0.0f);
        *reinterpret_cast<v4h*>(h16 + i) = o;
    }
}

// ---------------------------------------------------------------------------
// Fused GRU: one wave per 16-node tile, 8 column tiles, 6 WMMA accumulators
// per tile (i_r,i_i,i_n,h_r,h_i,h_n); gate chain in registers; hy -> out.
// Weights pre-transposed f16: Wxt/Wht [3D x D].
// ---------------------------------------------------------------------------
__global__ void gru_fused_kernel(float* __restrict__ out,
                                 const _Float16* __restrict__ h16,
                                 const _Float16* __restrict__ hid16,
                                 const float* __restrict__ hidden,
                                 const _Float16* __restrict__ Wxt,
                                 const float* __restrict__ bx,
                                 const _Float16* __restrict__ Wht,
                                 const float* __restrict__ bh) {
    const int D = 128;
    const int r0   = blockIdx.x * 16;
    const int lane = threadIdx.x & 31;
    const int half = lane >> 4;
    const int m    = lane & 15;
    const int n    = m;

    for (int j = 0; j < 8; ++j) {
        const int c0 = j * 16;
        v8f ixr = {}, ixi = {}, ixn = {};
        v8f hhr = {}, hhi = {}, hhn = {};
#pragma unroll 2
        for (int k0 = 0; k0 < D; k0 += 32) {
            v16h ah = load_a16(h16,   D, r0, k0, m, half);
            v16h av = load_a16(hid16, D, r0, k0, m, half);
            v16h bxr = load_bt16(Wxt, D, k0, c0,         n, half);
            v16h bxi = load_bt16(Wxt, D, k0, c0 + D,     n, half);
            v16h bxn = load_bt16(Wxt, D, k0, c0 + 2 * D, n, half);
            v16h bhr = load_bt16(Wht, D, k0, c0,         n, half);
            v16h bhi = load_bt16(Wht, D, k0, c0 + D,     n, half);
            v16h bhn = load_bt16(Wht, D, k0, c0 + 2 * D, n, half);
            ixr = wmma_f16(ah, bxr, ixr);
            ixi = wmma_f16(ah, bxi, ixi);
            ixn = wmma_f16(ah, bxn, ixn);
            hhr = wmma_f16(av, bhr, hhr);
            hhi = wmma_f16(av, bhi, hhi);
            hhn = wmma_f16(av, bhn, hhn);
        }
#pragma unroll
        for (int v = 0; v < 8; ++v) {
            int row = r0 + v + 8 * half;
            int col = c0 + n;
            float i_r = ixr[v] + bx[col];
            float i_i = ixi[v] + bx[D + col];
            float i_n = ixn[v] + bx[2 * D + col];
            float h_r = hhr[v] + bh[col];
            float h_i = hhi[v] + bh[D + col];
            float h_n = hhn[v] + bh[2 * D + col];
            float rg = 1.0f / (1.0f + expf(-(i_r + h_r)));
            float ig = 1.0f / (1.0f + expf(-(i_i + h_i)));
            float ng = tanhf(i_n + rg * h_n);
            float hv = hidden[(size_t)row * D + col];
            out[(size_t)row * D + col] = ng + ig * (hv - ng);
        }
    }
}

// ---------------------------------------------------------------------------
// Host launcher
// ---------------------------------------------------------------------------
extern "C" void kernel_launch(void* const* d_in, const int* in_sizes, int n_in,
                              void* d_out, int out_size, void* d_ws, size_t ws_size,
                              hipStream_t stream) {
    const int N = 50000, E = 800000, D = 128, L = 2, G = 3 * 128;

    const float* x       = (const float*)d_in[0];
    const float* hidden  = (const float*)d_in[1];
    const int*   adj_src = (const int*)  d_in[2];
    const int*   adj_dst = (const int*)  d_in[3];
    const float* adj_w   = (const float*)d_in[4];
    const float* gcn_W   = (const float*)d_in[5];
    const float* gcn_b   = (const float*)d_in[6];
    const float* x2h_W   = (const float*)d_in[7];
    const float* x2h_b   = (const float*)d_in[8];
    const float* h2h_W   = (const float*)d_in[9];
    const float* h2h_b   = (const float*)d_in[10];
    float*       out     = (float*)d_out;

    const size_t nd = (size_t)N * D;

    // Workspace layout:
    //   support f32[N*D] | agg f32[N*D]
    //   h16 f16[N*D] | hid16 f16[N*D]
    //   gcnWt16 f16[L*D*D] | x2hWt16 f16[G*D] | h2hWt16 f16[G*D]
    float*     support = (float*)d_ws;
    float*     agg     = support + nd;
    _Float16*  h16     = (_Float16*)(agg + nd);
    _Float16*  hid16   = h16 + nd;
    _Float16*  gcnWt16 = hid16 + nd;
    _Float16*  x2hWt16 = gcnWt16 + (size_t)L * D * D;
    _Float16*  h2hWt16 = x2hWt16 + (size_t)G * D;

    const int ndi = (int)nd;
    const long scat_threads = (long)E * 32;

    // One-time f16 prep
    cvt_f16_kernel<<<(ndi / 4 + 255) / 256, 256, 0, stream>>>(hidden, hid16, ndi);
    cvt_f16_kernel<<<(ndi / 4 + 255) / 256, 256, 0, stream>>>(x, h16, ndi);
    for (int l = 0; l < L; ++l) {
        transpose_cvt_kernel<<<(D * D + 255) / 256, 256, 0, stream>>>(
            gcn_W + (size_t)l * D * D, gcnWt16 + (size_t)l * D * D, D, D);
    }
    transpose_cvt_kernel<<<(D * G + 255) / 256, 256, 0, stream>>>(x2h_W, x2hWt16, D, G);
    transpose_cvt_kernel<<<(D * G + 255) / 256, 256, 0, stream>>>(h2h_W, h2hWt16, D, G);

    for (int l = 0; l < L; ++l) {
        // support = h16 @ gcn_W[l]   (Bt = transposed f16 weights)
        gemm_wmma_kernel<128><<<dim3(N / 16, D / 64), 32, 0, stream>>>(
            h16, gcnWt16 + (size_t)l * D * D, support, D);
        // agg = 0
        zero_kernel<<<(ndi + 255) / 256, 256, 0, stream>>>(agg, ndi);
        // agg[dst] += support[src] * w
        scatter_kernel<<<(unsigned)((scat_threads + 255) / 256), 256, 0, stream>>>(
            agg, support, adj_src + (size_t)l * E, adj_dst + (size_t)l * E,
            adj_w + (size_t)l * E, E);
        // h16 = f16(relu(agg + gcn_b[l]))  -- next layer / GRU input
        biasrelu_cvt_kernel<<<(ndi / 4 + 255) / 256, 256, 0, stream>>>(
            h16, agg, gcn_b + (size_t)l * D, ndi);
    }

    // Fused GRU gate chain -> out
    gru_fused_kernel<<<N / 16, 32, 0, stream>>>(out, h16, hid16, hidden,
                                                x2hWt16, x2h_b, h2hWt16, h2h_b);
}
